// EdgePredictor_70153995813491
// MI455X (gfx1250) — matrix operand order repaired and checked by reading, not compiled, not run on hardware
//
#include <hip/hip_runtime.h>
#include <hip/hip_bf16.h>

typedef __attribute__((ext_vector_type(16))) __bf16 v16bf;
typedef __attribute__((ext_vector_type(8)))  float  v8f;
typedef __attribute__((ext_vector_type(8)))  unsigned int v8u;

#define N_EDGES_C  500000
#define NODE_DIM_C 128
#define EDGE_DIM_C 64
#define IN_DIM_C   320
#define OUT_DIM_C  128
#define KT_C       10   // 320 / 32
#define NT_C       8    // 128 / 16
#define FRAGS_C    (KT_C * NT_C)        // 80 B-fragments
#define WSW_V8U    (FRAGS_C * 32)       // 2560 x 32B = 80 KB
#define WSW_BYTES  (WSW_V8U * 32)

// ---------------------------------------------------------------------------
// Prep: swizzle W (f32 [320][128]) into bf16 B-fragment order in d_ws.
// Fragment (kt,nt), lane: n = nt*16 + (lane&15), hi = lane>>4,
// element e (0..15): K = kt*32 + hi*16 + e   (contiguous k order; A uses the same)
// ---------------------------------------------------------------------------
__global__ void wprep_kernel(const float* __restrict__ W, v8u* __restrict__ wsw) {
  int tid = blockIdx.x * blockDim.x + threadIdx.x;
  if (tid >= WSW_V8U) return;
  int lane = tid & 31;
  int frag = tid >> 5;
  int kt = frag >> 3;
  int nt = frag & 7;
  int hi = lane >> 4;
  int n  = nt * 16 + (lane & 15);
  int kbase = kt * 32 + hi * 16;
  v16bf v;
#pragma unroll
  for (int e = 0; e < 16; ++e) {
    v[e] = (__bf16)W[(kbase + e) * OUT_DIM_C + n];
  }
  wsw[(size_t)frag * 32 + lane] = __builtin_bit_cast(v8u, v);
}

// ---------------------------------------------------------------------------
// Main: 8 waves/block, wave = 16 edges x 128 outputs. K-loop of 10 bf16 WMMAs
// per N-tile; A fragments gathered straight from global (L2-resident nodes).
// W staged into LDS via CDNA5 async global->LDS copy (ASYNCcnt tracked).
// ---------------------------------------------------------------------------
template<int O>
__device__ __forceinline__ void cvt4(v16bf& a, float4 f) {
  a[O + 0] = (__bf16)f.x; a[O + 1] = (__bf16)f.y;
  a[O + 2] = (__bf16)f.z; a[O + 3] = (__bf16)f.w;
}

__global__ __launch_bounds__(256, 1) void edge_mlp_kernel(
    const float* __restrict__ node,
    const float* __restrict__ edgef,
    const int*   __restrict__ srci,
    const int*   __restrict__ dsti,
    const float* __restrict__ bias,
    const v8u*   __restrict__ wsw,
    float*       __restrict__ out) {
  __shared__ v8u ldsW[WSW_V8U];   // 80 KB

  const int t = threadIdx.x;

  // ---- Async fill: global -> LDS directly, no VGPR round-trip. -----------
  // GVS addressing: mem = SADDR(64b SGPR) + VADDR(32b byte offset); dest LDS
  // address comes from the VDST VGPR (low 32 bits of the flat LDS pointer).
  {
    const unsigned long long gbase = (unsigned long long)(uintptr_t)wsw;
    const unsigned lds0 = (unsigned)(uintptr_t)(&ldsW[0]);
#pragma unroll
    for (int i = 0; i < WSW_BYTES / (256 * 16); ++i) {   // 20 x b128 per thread
      const unsigned byteoff = (unsigned)(t + i * 256) * 16u;
      const unsigned ldsaddr = lds0 + byteoff;
      asm volatile("global_load_async_to_lds_b128 %0, %1, %2 offset:0"
                   :: "v"(ldsaddr), "v"(byteoff), "s"(gbase)
                   : "memory");
    }
    asm volatile("s_wait_asynccnt 0" ::: "memory");
  }
  __syncthreads();

  const int wave = t >> 5;
  const int lane = t & 31;
  const int m16  = lane & 15;     // row (edge) within wave tile / column within C
  const int hi   = lane >> 4;     // K-half selector

  const long e0  = (long)blockIdx.x * 128 + (long)wave * 16;
  const long e   = e0 + m16;
  const long ecl = e < N_EDGES_C ? e : (N_EDGES_C - 1);

  const int s = srci[ecl];
  const int d = dsti[ecl];
  const float* psrc = node  + (long)s * NODE_DIM_C;
  const float* pdst = node  + (long)d * NODE_DIM_C;
  const float* pedg = edgef + ecl * EDGE_DIM_C;

  // Accumulators preloaded with bias (every row-VGPR of a tile shares n).
  v8f acc[NT_C];
#pragma unroll
  for (int nt = 0; nt < NT_C; ++nt) {
    const float bv = bias[nt * 16 + m16];
#pragma unroll
    for (int v = 0; v < 8; ++v) acc[nt][v] = bv;
  }

#pragma unroll
  for (int kt = 0; kt < KT_C; ++kt) {
    // Segment select resolves at compile time (loop fully unrolled); the
    // concat boundaries (128, 256) are multiples of 32 so runs never straddle.
    const float* base = (kt < 4) ? (psrc + kt * 32)
                      : (kt < 8) ? (pdst + (kt - 4) * 32)
                                 : (pedg + (kt - 8) * 32);
    const float4* p0 = (const float4*)(base + hi * 16);       // K = k0+hi*16 .. +7
    const float4* p1 = (const float4*)(base + hi * 16 + 8);   // K = k0+hi*16+8 .. +7
    const float4 f0 = p0[0], f1 = p0[1], f2 = p1[0], f3 = p1[1];

    v16bf a;
    cvt4<0>(a, f0);  cvt4<4>(a, f1);
    cvt4<8>(a, f2);  cvt4<12>(a, f3);

#pragma unroll
    for (int nt = 0; nt < NT_C; ++nt) {
      v16bf bfr = __builtin_bit_cast(v16bf, ldsW[(kt * NT_C + nt) * 32 + lane]);
      acc[nt] = __builtin_amdgcn_wmma_f32_16x16x32_bf16(
          false, a, false, bfr, (short)0, acc[nt], false, false);
    }
  }

  // Store: C/D layout -> lane holds column (nt*16 + m16), VGPR v holds row v + hi*8.
#pragma unroll
  for (int v = 0; v < 8; ++v) {
    const long row = e0 + v + hi * 8;
    if (row < N_EDGES_C) {
      float* orow = out + row * OUT_DIM_C + m16;
#pragma unroll
      for (int nt = 0; nt < NT_C; ++nt)
        orow[nt * 16] = acc[nt][v];
    }
  }
}

extern "C" void kernel_launch(void* const* d_in, const int* in_sizes, int n_in,
                              void* d_out, int out_size, void* d_ws, size_t ws_size,
                              hipStream_t stream) {
  const float* node  = (const float*)d_in[0];
  const float* edgef = (const float*)d_in[1];
  const int*   srci  = (const int*)  d_in[2];
  const int*   dsti  = (const int*)  d_in[3];
  const float* W     = (const float*)d_in[4];
  const float* b     = (const float*)d_in[5];
  float* out = (float*)d_out;
  v8u*   wsw = (v8u*)d_ws;   // needs 80 KB

  wprep_kernel<<<WSW_V8U / 256, 256, 0, stream>>>(W, wsw);

  const int nblocks = (N_EDGES_C + 127) / 128;   // 3907
  edge_mlp_kernel<<<nblocks, 256, 0, stream>>>(node, edgef, srci, dsti, b, wsw, out);
}
